// SpaGAT_48103633715624
// MI455X (gfx1250) — matrix operand order repaired and checked by reading, not compiled, not run on hardware
//
#include <hip/hip_runtime.h>
#include <hip/hip_bf16.h>
#include <math.h>

// ---------------------------------------------------------------------------
// Sparse GAT (pyGAT SpGraphAttentionLayer) forward for gfx1250 / MI455X.
//   L1: h = x@W1 (8 heads fused -> [N,512]), edge softmax-ish agg, elu
//   L2: h2 = h@W2 ([N,16]), edge agg, elu, log_softmax
// GEMMs use v_wmma_f32_16x16x32_bf16 with split-bf16 (hi+lo) inputs for
// near-fp32 accuracy on the fast bf16 matrix path.
// ---------------------------------------------------------------------------

namespace {
constexpr int kN   = 50000;
constexpr int kE   = 1600000;
constexpr int kF   = 512;          // NFEAT
constexpr int kHid = 64;           // NHID
constexpr int kH1  = 8;            // heads layer 1
constexpr int kC1  = kH1 * kHid;   // 512 = concat width (== NFEAT, convenient)
constexpr int kNC  = 16;           // NCLASS
constexpr float kAlpha = 0.2f;     // leaky_relu slope
}

typedef __bf16 bf16;
typedef __attribute__((ext_vector_type(16))) __bf16 v16bf;
typedef __attribute__((ext_vector_type(8)))  __bf16 bf16x8;
typedef __attribute__((ext_vector_type(8)))  float  v8f;

__device__ __forceinline__ void atomAddF(float* p, float v) {
  __hip_atomic_fetch_add(p, v, __ATOMIC_RELAXED, __HIP_MEMORY_SCOPE_AGENT);
}

// A-fragment (16x32 bf16, row-major source): lane holds row (lane&15),
// K = k0..k0+7 and k0+16..k0+23 with k0 = (lane>>4)*8. Two 16B loads.
__device__ __forceinline__ v16bf load_a_frag(const bf16* __restrict__ p) {
  bf16x8 x0 = *(const bf16x8*)(p);
  bf16x8 x1 = *(const bf16x8*)(p + 16);
  return __builtin_shufflevector(x0, x1, 0,1,2,3,4,5,6,7,8,9,10,11,12,13,14,15);
}
// B-fragment (32x16 bf16, from transposed Bt[n][k]): lane holds column
// (lane&15), K = k0..k0+7 and k0+16..k0+23 with k0 = (lane>>4)*8. Two 16B loads.
__device__ __forceinline__ v16bf load_b_frag(const bf16* __restrict__ p) {
  bf16x8 x0 = *(const bf16x8*)(p);
  bf16x8 x1 = *(const bf16x8*)(p + 16);
  return __builtin_shufflevector(x0, x1, 0,1,2,3,4,5,6,7,8,9,10,11,12,13,14,15);
}

// f32 -> (bf16 hi, bf16 lo) split conversion
__global__ __launch_bounds__(256) void k_split_bf16(const float* __restrict__ in,
                                                    bf16* __restrict__ hi,
                                                    bf16* __restrict__ lo, int n) {
  int t = blockIdx.x * 256 + threadIdx.x;
  if (t >= n) return;
  float v = in[t];
  bf16 h = (bf16)v;
  hi[t] = h;
  lo[t] = (bf16)(v - (float)h);
}

// W1[H1][F][HID] -> Bt1[n=head*64+f][k] (bf16 hi/lo), k in [0,512)
__global__ __launch_bounds__(256) void k_prep_B1(const float* __restrict__ W1,
                                                 bf16* __restrict__ hi,
                                                 bf16* __restrict__ lo) {
  int t = blockIdx.x * 256 + threadIdx.x;
  if (t >= kC1 * kF) return;
  int n = t >> 9, k = t & 511;
  int head = n >> 6, f = n & 63;
  float v = W1[((size_t)head * kF + k) * kHid + f];
  bf16 h = (bf16)v;
  hi[t] = h;
  lo[t] = (bf16)(v - (float)h);
}

// W2[1][512][16] -> Bt2[c][k]
__global__ __launch_bounds__(256) void k_prep_B2(const float* __restrict__ W2,
                                                 bf16* __restrict__ hi,
                                                 bf16* __restrict__ lo) {
  int t = blockIdx.x * 256 + threadIdx.x;
  if (t >= kNC * kF) return;
  int c = t >> 9, k = t & 511;
  float v = W2[(size_t)k * kNC + c];
  bf16 h = (bf16)v;
  hi[t] = h;
  lo[t] = (bf16)(v - (float)h);
}

// Split-bf16 GEMM: C[M,ldc] = A[M,K] * B[K,Ncols], B given transposed (Bt[n][k]).
// One 16x16 output tile per wave; 8 waves/block; consecutive waves share the
// same M tile (A reuse in L0/L2). 3 WMMAs per K-step: hi*hi + lo*hi + hi*lo.
__global__ __launch_bounds__(256) void k_gemm_split(
    const bf16* __restrict__ Ahi, const bf16* __restrict__ Alo,
    const bf16* __restrict__ Bhi, const bf16* __restrict__ Blo,
    float* __restrict__ C, int mTiles, int nTiles, int K, int ldc) {
  const int wave = threadIdx.x >> 5, lane = threadIdx.x & 31;
  const int gw = blockIdx.x * 8 + wave;
  if (gw >= mTiles * nTiles) return;           // uniform per wave: EXEC stays full
  const int mT = gw / nTiles, nT = gw - mT * nTiles;
  const int half = lane >> 4, l15 = lane & 15;
  const size_t aOff = (size_t)(mT * 16 + l15) * K + half * 8;
  const size_t bOff = (size_t)(nT * 16 + l15) * K + half * 8;
  v8f acc = {};
  for (int k = 0; k < K; k += 32) {
    if (k + 32 < K) {                          // global_prefetch_b8 of next K block
      __builtin_prefetch(Ahi + aOff + k + 32, 0, 3);
      __builtin_prefetch(Bhi + bOff + k + 32, 0, 3);
    }
    v16bf ah = load_a_frag(Ahi + aOff + k);
    v16bf al = load_a_frag(Alo + aOff + k);
    v16bf bh = load_b_frag(Bhi + bOff + k);
    v16bf bl = load_b_frag(Blo + bOff + k);
    acc = __builtin_amdgcn_wmma_f32_16x16x32_bf16(false, ah, false, bh, (short)0, acc, false, false);
    acc = __builtin_amdgcn_wmma_f32_16x16x32_bf16(false, al, false, bh, (short)0, acc, false, false);
    acc = __builtin_amdgcn_wmma_f32_16x16x32_bf16(false, ah, false, bl, (short)0, acc, false, false);
  }
  // C layout: VGPR r -> row m0 + (lane>>4)*8 + r, col n0 + (lane&15)
  float* Cp = C + (size_t)(mT * 16 + half * 8) * ldc + (size_t)nT * 16 + l15;
#pragma unroll
  for (int r = 0; r < 8; ++r) Cp[(size_t)r * ldc] = acc[r];
}

// Per-(node, head) attention scores: s_src = h.a[:64], s_dst = h.a[64:]
__global__ __launch_bounds__(256) void k_scores1(const float* __restrict__ hlin,
                                                 const float* __restrict__ a1,
                                                 float* __restrict__ s1) {
  int n = blockIdx.x * 256 + threadIdx.x;
  int h = blockIdx.y;
  if (n >= kN) return;
  const float* hv = hlin + (size_t)n * kC1 + h * kHid;
  const float* av = a1 + (size_t)h * 2 * kHid;
  float ss = 0.f, sd = 0.f;
#pragma unroll 8
  for (int f = 0; f < kHid; ++f) {
    float x = hv[f];
    ss += x * av[f];
    sd += x * av[kHid + f];
  }
  s1[(size_t)h * kN + n] = ss;
  s1[(size_t)(kH1 + h) * kN + n] = sd;
}

// Layer-1 edge pass: one wave per (edge, head); each lane handles 2 features.
__global__ __launch_bounds__(256) void k_edge1(const int* __restrict__ ei,
                                               const float* __restrict__ hlin,
                                               const float* __restrict__ s1,
                                               float* __restrict__ rowsum,
                                               float* __restrict__ agg) {
  const int wave = threadIdx.x >> 5, lane = threadIdx.x & 31;
  const long long gw = (long long)blockIdx.x * 8 + wave;
  if (gw >= (long long)kE * kH1) return;
  const int e = (int)(gw >> 3), h = (int)(gw & 7);
  const int s = ei[e], d = ei[kE + e];
  float lv = s1[(size_t)h * kN + s] + s1[(size_t)(kH1 + h) * kN + d];
  float lr = lv > 0.f ? lv : kAlpha * lv;
  float wgt = expf(-lr);
  if (lane == 0) atomAddF(rowsum + (size_t)h * kN + s, wgt);
  const float* hv = hlin + (size_t)d * kC1 + h * kHid;
  float*       av = agg + (size_t)s * kC1 + h * kHid;
  float2 x = *(const float2*)(hv + lane * 2);
  atomAddF(av + lane * 2,     wgt * x.x);
  atomAddF(av + lane * 2 + 1, wgt * x.y);
}

// h1 = elu(agg/rowsum), written as split bf16 (input of GEMM2)
__global__ __launch_bounds__(256) void k_fin1(const float* __restrict__ agg,
                                              const float* __restrict__ rowsum,
                                              bf16* __restrict__ hi,
                                              bf16* __restrict__ lo) {
  int t = blockIdx.x * 256 + threadIdx.x;
  if (t >= kN * kC1) return;
  int n = t >> 9, c = t & 511, h = c >> 6;
  float v = agg[t] / rowsum[(size_t)h * kN + n];
  float r = v > 0.f ? v : expm1f(v);
  bf16 hb = (bf16)r;
  hi[t] = hb;
  lo[t] = (bf16)(r - (float)hb);
}

__global__ __launch_bounds__(256) void k_scores2(const float* __restrict__ hlin2,
                                                 const float* __restrict__ a2,
                                                 float* __restrict__ s2) {
  int n = blockIdx.x * 256 + threadIdx.x;
  if (n >= kN) return;
  float ss = 0.f, sd = 0.f;
#pragma unroll
  for (int c = 0; c < kNC; ++c) {
    float x = hlin2[(size_t)n * kNC + c];
    ss += x * a2[c];
    sd += x * a2[kNC + c];
  }
  s2[n] = ss;
  s2[kN + n] = sd;
}

// Layer-2 edge pass: thread per (edge, class)
__global__ __launch_bounds__(256) void k_edge2(const int* __restrict__ ei,
                                               const float* __restrict__ hlin2,
                                               const float* __restrict__ s2,
                                               float* __restrict__ rowsum2,
                                               float* __restrict__ agg2) {
  int t = blockIdx.x * 256 + threadIdx.x;
  if (t >= kE * kNC) return;
  int e = t >> 4, f = t & 15;
  int s = ei[e], d = ei[kE + e];
  float lv = s2[s] + s2[kN + d];
  float lr = lv > 0.f ? lv : kAlpha * lv;
  float wgt = expf(-lr);
  if (f == 0) atomAddF(rowsum2 + s, wgt);
  atomAddF(agg2 + (size_t)s * kNC + f, wgt * hlin2[(size_t)d * kNC + f]);
}

// out = log_softmax(elu(agg2/rowsum2)) over 16 classes
__global__ __launch_bounds__(256) void k_final(const float* __restrict__ agg2,
                                               const float* __restrict__ rowsum2,
                                               float* __restrict__ out) {
  int n = blockIdx.x * 256 + threadIdx.x;
  if (n >= kN) return;
  float r = rowsum2[n];
  float v[kNC];
  float m = -INFINITY;
#pragma unroll
  for (int c = 0; c < kNC; ++c) {
    float x = agg2[(size_t)n * kNC + c] / r;
    x = x > 0.f ? x : expm1f(x);
    v[c] = x;
    m = fmaxf(m, x);
  }
  float sum = 0.f;
#pragma unroll
  for (int c = 0; c < kNC; ++c) sum += expf(v[c] - m);
  float l = m + logf(sum);
#pragma unroll
  for (int c = 0; c < kNC; ++c) out[(size_t)n * kNC + c] = v[c] - l;
}

extern "C" void kernel_launch(void* const* d_in, const int* in_sizes, int n_in,
                              void* d_out, int out_size, void* d_ws, size_t ws_size,
                              hipStream_t stream) {
  (void)in_sizes; (void)n_in; (void)out_size; (void)ws_size;
  const float* x  = (const float*)d_in[0];
  const int*   ei = (const int*)d_in[1];
  const float* W1 = (const float*)d_in[2];
  const float* a1 = (const float*)d_in[3];
  const float* W2 = (const float*)d_in[4];
  const float* a2 = (const float*)d_in[5];
  float* out = (float*)d_out;

  char* w = (char*)d_ws;
  size_t off = 0;
  auto alloc = [&](size_t bytes) {
    size_t a = off;
    off += (bytes + 255) & ~(size_t)255;
    return a;
  };
  // accumulators placed contiguously -> one memset per call
  const size_t oXhi   = alloc((size_t)kN * kC1 * 2);   // bf16 hi of x / h1
  const size_t oXlo   = alloc((size_t)kN * kC1 * 2);   // bf16 lo
  const size_t oHlin1 = alloc((size_t)kN * kC1 * 4);   // x@W1 (all heads)
  const size_t oAgg1  = alloc((size_t)kN * kC1 * 4);   // zeroed
  const size_t oRow1  = alloc((size_t)kH1 * kN * 4);   // zeroed
  const size_t oAgg2  = alloc((size_t)kN * kNC * 4);   // zeroed
  const size_t oRow2  = alloc((size_t)kN * 4);         // zeroed
  const size_t oHlin2 = alloc((size_t)kN * kNC * 4);
  const size_t oS1    = alloc((size_t)2 * kH1 * kN * 4);
  const size_t oS2    = alloc((size_t)2 * kN * 4);
  const size_t oB1hi  = alloc((size_t)kC1 * kF * 2);
  const size_t oB1lo  = alloc((size_t)kC1 * kF * 2);
  const size_t oB2hi  = alloc((size_t)kNC * kF * 2);
  const size_t oB2lo  = alloc((size_t)kNC * kF * 2);

  bf16*  Xhi   = (bf16*)(w + oXhi);
  bf16*  Xlo   = (bf16*)(w + oXlo);
  float* Hlin1 = (float*)(w + oHlin1);
  float* Agg1  = (float*)(w + oAgg1);
  float* Row1  = (float*)(w + oRow1);
  float* Agg2  = (float*)(w + oAgg2);
  float* Row2  = (float*)(w + oRow2);
  float* Hlin2 = (float*)(w + oHlin2);
  float* S1    = (float*)(w + oS1);
  float* S2    = (float*)(w + oS2);
  bf16*  B1hi  = (bf16*)(w + oB1hi);
  bf16*  B1lo  = (bf16*)(w + oB1lo);
  bf16*  B2hi  = (bf16*)(w + oB2hi);
  bf16*  B2lo  = (bf16*)(w + oB2lo);
  (void)Row1;

  // zero all accumulators (agg1, row1, agg2, row2 are contiguous)
  hipMemsetAsync(w + oAgg1, 0, oHlin2 - oAgg1, stream);

  const int elemsX = kN * kC1;                          // 25.6M
  k_split_bf16<<<(elemsX + 255) / 256, 256, 0, stream>>>(x, Xhi, Xlo, elemsX);
  k_prep_B1<<<(kC1 * kF + 255) / 256, 256, 0, stream>>>(W1, B1hi, B1lo);
  k_prep_B2<<<(kNC * kF + 255) / 256, 256, 0, stream>>>(W2, B2hi, B2lo);

  // GEMM1: [50000,512] x [512,512] ; tiles = 3125 * 32 = 100000, 8 waves/block
  const int mTiles = kN / 16;                           // 3125
  const int nT1 = kC1 / 16;                             // 32
  k_gemm_split<<<(mTiles * nT1 + 7) / 8, 256, 0, stream>>>(
      Xhi, Xlo, B1hi, B1lo, Hlin1, mTiles, nT1, kF, kC1);

  k_scores1<<<dim3((kN + 255) / 256, kH1), 256, 0, stream>>>(Hlin1, a1, S1);
  k_edge1<<<(int)(((long long)kE * kH1) / 8), 256, 0, stream>>>(ei, Hlin1, S1, Row1, Agg1);
  k_fin1<<<(elemsX + 255) / 256, 256, 0, stream>>>(Agg1, Row1, Xhi, Xlo);

  // GEMM2: [50000,512] x [512,16]
  k_gemm_split<<<(mTiles * 1 + 7) / 8, 256, 0, stream>>>(
      Xhi, Xlo, B2hi, B2lo, Hlin2, mTiles, 1, kC1, kNC);

  k_scores2<<<(kN + 255) / 256, 256, 0, stream>>>(Hlin2, a2, S2);
  k_edge2<<<(kE * kNC + 255) / 256, 256, 0, stream>>>(ei, Hlin2, S2, Row2, Agg2);
  k_final<<<(kN + 255) / 256, 256, 0, stream>>>(Agg2, Row2, out);
}
